// LA_87488483820014
// MI455X (gfx1250) — matrix operand order repaired and checked by reading, not compiled, not run on hardware
//
#include <hip/hip_runtime.h>

// Problem constants (match reference)
#define NNODES 50000
#define FIN    256
#define NH     4
#define NC     64
#define NHC    256
#define NE     800000
#define KPAD   50016   // NNODES padded to multiple of 32 for WMMA K-loop
#define GSPLIT 16      // split-K factor for the Gram GEMM
#define GSTEPS 98      // ceil(1563/16) k-steps of 32 per split

typedef __attribute__((ext_vector_type(16))) __bf16 v16bf;
typedef __attribute__((ext_vector_type(8)))  float  v8f;
typedef __attribute__((ext_vector_type(4)))  unsigned int v4u;
typedef __attribute__((ext_vector_type(8)))  int    v8i;
typedef __attribute__((ext_vector_type(4)))  int    v4i;

#if defined(__has_builtin)
#if __has_builtin(__builtin_amdgcn_tensor_load_to_lds) && __has_builtin(__builtin_amdgcn_s_wait_tensorcnt)
#define HAVE_TDM 1
#endif
#endif

union FragAB { uint4 u[2]; v16bf v; };

__device__ __forceinline__ unsigned short f2bf(float f) {
  unsigned int u = __float_as_uint(f);
  u += 0x7FFFu + ((u >> 16) & 1u);          // round-to-nearest-even
  return (unsigned short)(u >> 16);
}

__device__ __forceinline__ void atomicMaxF(float* addr, float val) {
  // float ordering == int ordering for >=0, reversed-unsigned for <0
  if (val >= 0.0f) atomicMax((int*)addr, __float_as_int(val));
  else             atomicMin((unsigned int*)addr, __float_as_uint(val));
}

// ---------------- casts / packing ----------------

__global__ void cast_x_kernel(const float* __restrict__ x, unsigned short* __restrict__ xb) {
  int tid = blockIdx.x * 256 + threadIdx.x;
  if (tid < NNODES * FIN) xb[tid] = f2bf(x[tid]);
}

__global__ void prep_w_kernel(const float* __restrict__ Wq, const float* __restrict__ Wk,
                              const float* __restrict__ Wv, const float* __restrict__ Ws,
                              const float* __restrict__ bq, const float* __restrict__ bk,
                              const float* __restrict__ bv, const float* __restrict__ bs,
                              unsigned short* __restrict__ Wt, float* __restrict__ b4) {
  int tid = blockIdx.x * 256 + threadIdx.x;   // 0 .. 1024*256-1
  int col = tid >> 8;                          // 0..1023 (q|k|v|skip)
  int k   = tid & 255;
  int sel = col >> 8;
  int c   = col & 255;
  const float* W = (sel == 0) ? Wq : (sel == 1) ? Wk : (sel == 2) ? Wv : Ws;
  Wt[col * 256 + k] = f2bf(W[k * 256 + c]);    // column-major pack: Wt[col][k]
  if (k == 0) {
    const float* B = (sel == 0) ? bq : (sel == 1) ? bk : (sel == 2) ? bv : bs;
    b4[col] = B[c];
  }
}

// ---------------- fused QKV+skip projection: P = X @ [Wq|Wk|Wv|Wskip] + b ----------------
// One wave per 16x16 tile; v_wmma_f32_16x16x32_bf16, K = 256 in 8 steps.
// The block's 64KB weight slice (128 contiguous columns of Wt) is staged into LDS
// by the Tensor Data Mover (one tensor_load_to_lds per block), then B-fragments
// are read from LDS (ds_load_b128).

__global__ void gemm_proj_kernel(const unsigned short* __restrict__ xb,
                                 const unsigned short* __restrict__ Wt,
                                 const float* __restrict__ b4,
                                 float* __restrict__ P,       // [N][768]  (Q|K|V)
                                 float* __restrict__ outb) {  // [N][256]  (skip part)
  __shared__ unsigned short shmB[128 * 256];   // 64 KB: 128 cols x 256 K (col-major)

  const int lane = threadIdx.x & 31;
  const int wave = threadIdx.x >> 5;
  const int rowBase = blockIdx.x * 16;
  const int colTile = blockIdx.y * 8 + wave;    // 0..63
  const int colBase = colTile * 16;
  const int half = lane >> 4;                   // 0 | 1
  const int l16  = lane & 15;

#ifdef HAVE_TDM
  if (threadIdx.x == 0) {
    // D# per CDNA5 ISA ch.8: 1-D contiguous tile, 8192 x 8B = 64KB
    unsigned long long gaddr = (unsigned long long)(Wt + (size_t)blockIdx.y * 128 * 256);
    unsigned ldsoff = (unsigned)(unsigned long long)(&shmB[0]); // low 32b of flat LDS addr
    v4u g0;
    g0[0] = 1u;                                           // count=1, user mode
    g0[1] = ldsoff;                                       // lds_addr
    g0[2] = (unsigned)(gaddr & 0xFFFFFFFFu);              // global_addr[31:0]
    g0[3] = (unsigned)((gaddr >> 32) & 0x01FFFFFFu) | (2u << 30); // addr[56:32] | type=2
    v8i g1;
    g1[0] = (int)(3u << 16);          // data_size = 8B
    g1[1] = (int)(8192u << 16);       // tensor_dim0[15:0]=8192
    g1[2] = (int)(1u << 16);          // tensor_dim0[31:16]=0, tensor_dim1=1
    g1[3] = (int)(8192u << 16);       // tile_dim0 = 8192
    g1[4] = 0;                        // tile_dim1/2 unused
    g1[5] = 8192;                     // tensor_dim0_stride = 8192
    g1[6] = 0; g1[7] = 0;
    v4i g2 = {0, 0, 0, 0}, g3 = {0, 0, 0, 0};
#if defined(__clang_major__) && (__clang_major__ >= 23)
    v8i g4 = {0, 0, 0, 0, 0, 0, 0, 0};
    __builtin_amdgcn_tensor_load_to_lds(g0, g1, g2, g3, g4, 0);
#else
    __builtin_amdgcn_tensor_load_to_lds(g0, g1, g2, g3, 0);
#endif
    __builtin_amdgcn_s_wait_tensorcnt(0);
  }
  __syncthreads();
#else
  // Fallback: cooperative global->LDS copy (128 cols x 512B, contiguous 64KB)
  {
    const uint4* src = (const uint4*)(Wt + (size_t)blockIdx.y * 128 * 256);
    uint4* dst = (uint4*)&shmB[0];
    for (int i = threadIdx.x; i < 4096; i += 256) dst[i] = src[i];
  }
  __syncthreads();
#endif

  // A: lane<16 -> row l16, K {0..7,16..23}; lane>=16 -> K {8..15,24..31}
  const unsigned short* aptr = xb + (size_t)(rowBase + l16) * FIN + half * 8;
  // B from LDS: lane<16 -> col l16, K {0..7,16..23}; lane>=16 -> K {8..15,24..31}
  const unsigned short* bptr = &shmB[(wave * 16 + l16) * 256 + half * 8];

  v8f c = {};
#pragma unroll
  for (int k0 = 0; k0 < FIN; k0 += 32) {
    FragAB a, b;
    a.u[0] = *(const uint4*)(aptr + k0);
    a.u[1] = *(const uint4*)(aptr + k0 + 16);
    b.u[0] = *(const uint4*)(bptr + k0);
    b.u[1] = *(const uint4*)(bptr + k0 + 16);
    c = __builtin_amdgcn_wmma_f32_16x16x32_bf16(false, a.v, false, b.v,
                                                (short)0, c, false, false);
  }

  const int col = colBase + l16;
  const float bias = b4[col];
#pragma unroll
  for (int r = 0; r < 8; ++r) {
    int row = rowBase + r + half * 8;
    float val = c[r] + bias;
    if (col < 768) P[(size_t)row * 768 + col] = val;
    else           outb[(size_t)row * NHC + (col - 768)] = val;
  }
}

// ---------------- edge attention ----------------

__global__ void init_stats_kernel(float* __restrict__ amax, float* __restrict__ denom) {
  int tid = blockIdx.x * 256 + threadIdx.x;
  if (tid < NNODES * NH) { amax[tid] = -3.0e38f; denom[tid] = 0.0f; }
}

// one block per edge: 256 threads = 256 channels; coalesced Q/K loads,
// LDS tree-reduction per 64-channel head segment.
__global__ void edge_alpha_kernel(const long long* __restrict__ ei,
                                  const float* __restrict__ eattr,
                                  const float* __restrict__ We, const float* __restrict__ be,
                                  const float* __restrict__ P,
                                  float* __restrict__ alpha, float* __restrict__ amax) {
  __shared__ float red[256];
  int e  = blockIdx.x;
  int hc = threadIdx.x;
  int h  = hc >> 6;
  int src = (int)ei[e], dst = (int)ei[NE + e];
  float ea = eattr[e];
  float q = P[(size_t)dst * 768 + hc];
  float k = P[(size_t)src * 768 + 256 + hc];
  red[hc] = q * (k + ea * We[hc] + be[hc]);
  __syncthreads();
#pragma unroll
  for (int off = 32; off > 0; off >>= 1) {
    if ((hc & 63) < off) red[hc] += red[hc + off];
    __syncthreads();
  }
  if ((hc & 63) == 0) {
    float a = red[hc] * 0.125f;       // 1/sqrt(C=64)
    alpha[e * NH + h] = a;
    atomicMaxF(&amax[dst * NH + h], a);
  }
}

__global__ void edge_ex_kernel(const long long* __restrict__ ei,
                               float* __restrict__ alpha,
                               const float* __restrict__ amax,
                               float* __restrict__ denom) {
  int tid = blockIdx.x * 256 + threadIdx.x;
  if (tid >= NE * NH) return;
  int e = tid >> 2, h = tid & 3;
  int dst = (int)ei[NE + e];
  float ex = __expf(alpha[tid] - amax[dst * NH + h]);
  alpha[tid] = ex;
  atomicAdd(&denom[dst * NH + h], ex);
}

// one block per edge: one coalesced atomicAdd per channel
__global__ void edge_agg_kernel(const long long* __restrict__ ei,
                                const float* __restrict__ eattr,
                                const float* __restrict__ We, const float* __restrict__ be,
                                const float* __restrict__ P,
                                const float* __restrict__ alpha,
                                const float* __restrict__ denom,
                                float* __restrict__ outb) {
  int e  = blockIdx.x;
  int hc = threadIdx.x;
  int h  = hc >> 6;
  int src = (int)ei[e], dst = (int)ei[NE + e];
  float attn = alpha[e * NH + h] / (denom[dst * NH + h] + 1e-16f);
  float ea = eattr[e];
  float v = P[(size_t)src * 768 + 512 + hc];
  atomicAdd(&outb[(size_t)dst * NHC + hc], (v + ea * We[hc] + be[hc]) * attn);
}

// ---------------- GraphNorm ----------------

__global__ void col_stats_kernel(const float* __restrict__ outb,
                                 const float* __restrict__ mscale,
                                 float* __restrict__ meanv, float* __restrict__ varv) {
  __shared__ float ssum[256], ssq[256];
  int hc = blockIdx.x, t = threadIdx.x;
  float s = 0.0f, q = 0.0f;
  for (int i = t; i < NNODES; i += 256) {
    float v = outb[(size_t)i * NHC + hc];
    s += v; q += v * v;
  }
  ssum[t] = s; ssq[t] = q;
  __syncthreads();
  for (int off = 128; off > 0; off >>= 1) {
    if (t < off) { ssum[t] += ssum[t + off]; ssq[t] += ssq[t + off]; }
    __syncthreads();
  }
  if (t == 0) {
    float m  = ssum[0] / (float)NNODES;
    float ms = mscale[hc];
    // E[(out - ms*m)^2] = E[out^2] - 2*ms*m*E[out] + ms^2*m^2
    float var = ssq[0] / (float)NNODES - 2.0f * ms * m * m + ms * ms * m * m;
    meanv[hc] = m; varv[hc] = var;
  }
}

// normalize + ReLU, write transposed bf16 Yt[256][KPAD] (pad region zeroed)
__global__ void norm_relu_t_kernel(const float* __restrict__ outb,
                                   const float* __restrict__ meanv, const float* __restrict__ varv,
                                   const float* __restrict__ mscale,
                                   const float* __restrict__ gw, const float* __restrict__ gb,
                                   unsigned short* __restrict__ Yt) {
  int hc = blockIdx.y;
  int i  = blockIdx.x * 256 + threadIdx.x;
  if (i >= KPAD) return;
  float val = 0.0f;
  if (i < NNODES) {
    float v = outb[(size_t)i * NHC + hc];
    v = (v - mscale[hc] * meanv[hc]) * rsqrtf(varv[hc] + 1e-5f) * gw[hc] + gb[hc];
    val = fmaxf(v, 0.0f);
  }
  Yt[(size_t)hc * KPAD + i] = f2bf(val);
}

// ---------------- Gram matrix: xt = Y^T @ Y via WMMA, split-K ----------------

__global__ void zero_xt_kernel(float* __restrict__ xt) {
  int tid = blockIdx.x * 256 + threadIdx.x;
  if (tid < NHC * NHC) xt[tid] = 0.0f;
}

__global__ void gram_wmma_kernel(const unsigned short* __restrict__ Yt,
                                 float* __restrict__ xt) {
  const int lane = threadIdx.x & 31;
  const int wave = threadIdx.x >> 5;
  const int tile = blockIdx.x * 8 + wave;       // 0..255
  const int rowBase = (tile >> 4) * 16;
  const int colBase = (tile & 15) * 16;
  const int half = lane >> 4;
  const int l16  = lane & 15;

  const int kBegin = blockIdx.y * (GSTEPS * 32);
  const int kEnd   = min(KPAD, kBegin + GSTEPS * 32);

  const unsigned short* aptr = Yt + (size_t)(rowBase + l16) * KPAD + half * 8;
  const unsigned short* bptr = Yt + (size_t)(colBase + l16) * KPAD + half * 8;

  v8f c = {};
  for (int k0 = kBegin; k0 < kEnd; k0 += 32) {
    __builtin_prefetch(aptr + k0 + 256, 0, 0);   // global_prefetch ahead
    __builtin_prefetch(bptr + k0 + 256, 0, 0);
    FragAB a, b;
    a.u[0] = *(const uint4*)(aptr + k0);
    a.u[1] = *(const uint4*)(aptr + k0 + 16);
    b.u[0] = *(const uint4*)(bptr + k0);
    b.u[1] = *(const uint4*)(bptr + k0 + 16);
    c = __builtin_amdgcn_wmma_f32_16x16x32_bf16(false, a.v, false, b.v,
                                                (short)0, c, false, false);
  }
#pragma unroll
  for (int r = 0; r < 8; ++r) {
    int row = rowBase + r + half * 8;
    atomicAdd(&xt[row * NHC + colBase + l16], c[r]);
  }
}

__global__ void minmax_norm_kernel(float* __restrict__ xt) {
  __shared__ float smin[256], smax[256];
  int t = threadIdx.x;
  float mn = 3.0e38f, mx = -3.0e38f;
  for (int i = t; i < NHC * NHC; i += 256) {
    float v = xt[i];
    mn = fminf(mn, v); mx = fmaxf(mx, v);
  }
  smin[t] = mn; smax[t] = mx;
  __syncthreads();
  for (int off = 128; off > 0; off >>= 1) {
    if (t < off) { smin[t] = fminf(smin[t], smin[t + off]); smax[t] = fmaxf(smax[t], smax[t + off]); }
    __syncthreads();
  }
  float lo = smin[0];
  float inv = 1.0f / (smax[0] - lo + 1e-8f);
  for (int i = t; i < NHC * NHC; i += 256) xt[i] = (xt[i] - lo) * inv;
}

// ---------------- launch ----------------

extern "C" void kernel_launch(void* const* d_in, const int* in_sizes, int n_in,
                              void* d_out, int out_size, void* d_ws, size_t ws_size,
                              hipStream_t stream) {
  (void)in_sizes; (void)n_in; (void)out_size; (void)ws_size;

  const float*     x     = (const float*)d_in[0];
  const long long* ei    = (const long long*)d_in[1];
  const float*     eattr = (const float*)d_in[2];
  const float*     Wq    = (const float*)d_in[3];
  const float*     bq    = (const float*)d_in[4];
  const float*     Wk    = (const float*)d_in[5];
  const float*     bk    = (const float*)d_in[6];
  const float*     Wv    = (const float*)d_in[7];
  const float*     bv    = (const float*)d_in[8];
  const float*     We    = (const float*)d_in[9];
  const float*     be    = (const float*)d_in[10];
  const float*     Ws    = (const float*)d_in[11];
  const float*     bs    = (const float*)d_in[12];
  const float*     gw    = (const float*)d_in[13];
  const float*     gb    = (const float*)d_in[14];
  const float*     gms   = (const float*)d_in[15];

  char* ws = (char*)d_ws;
  unsigned short* xb    = (unsigned short*)(ws + 0);                 //  25,600,000 B
  unsigned short* Wt    = (unsigned short*)(ws + 25600000);          //     524,288 B
  float*          b4    = (float*)(ws + 26124288);                   //       4,096 B
  float*          P     = (float*)(ws + 26128384);                   // 153,600,000 B  [N][768] Q|K|V
  float*          outb  = (float*)(ws + 179728384);                  //  51,200,000 B  [N][256]
  float*          alpha = (float*)(ws + 230928384);                  //  12,800,000 B  [E][4]
  float*          amax  = (float*)(ws + 243728384);                  //     800,000 B
  float*          denom = (float*)(ws + 244528384);                  //     800,000 B
  float*          meanv = (float*)(ws + 245328384);                  //       1,024 B
  float*          varv  = (float*)(ws + 245329408);                  //       1,024 B
  unsigned short* Yt    = (unsigned short*)(ws + 245330432);         //  25,608,192 B  [256][KPAD]
  float*          xt    = (float*)d_out;

  cast_x_kernel<<<(NNODES * FIN + 255) / 256, 256, 0, stream>>>(x, xb);
  prep_w_kernel<<<(1024 * 256) / 256, 256, 0, stream>>>(Wq, Wk, Wv, Ws, bq, bk, bv, bs, Wt, b4);
  gemm_proj_kernel<<<dim3(NNODES / 16, 8), 256, 0, stream>>>(xb, Wt, b4, P, outb);
  init_stats_kernel<<<(NNODES * NH + 255) / 256, 256, 0, stream>>>(amax, denom);
  edge_alpha_kernel<<<NE, 256, 0, stream>>>(ei, eattr, We, be, P, alpha, amax);
  edge_ex_kernel<<<(NE * NH + 255) / 256, 256, 0, stream>>>(ei, alpha, amax, denom);
  edge_agg_kernel<<<NE, 256, 0, stream>>>(ei, eattr, We, be, P, alpha, denom, outb);
  col_stats_kernel<<<NHC, 256, 0, stream>>>(outb, gms, meanv, varv);
  norm_relu_t_kernel<<<dim3((KPAD + 255) / 256, NHC), 256, 0, stream>>>(outb, meanv, varv, gms, gw, gb, Yt);
  zero_xt_kernel<<<(NHC * NHC) / 256, 256, 0, stream>>>(xt);
  gram_wmma_kernel<<<dim3(32, GSPLIT), 256, 0, stream>>>(Yt, xt);
  minmax_norm_kernel<<<1, 256, 0, stream>>>(xt);
}
